// GATDecoder_39565238731347
// MI455X (gfx1250) — matrix operand order repaired and checked by reading, not compiled, not run on hardware
//
#include <hip/hip_runtime.h>

// ---------------- problem constants ----------------
#define B_    16
#define LAT_  128
#define HID_  256
#define N_    256
#define H_    4
#define C_    64
#define NHID_ (N_ * HID_)   // 65536

// ---------------- WMMA helpers (gfx1250, wave32) ----------------
typedef __attribute__((ext_vector_type(16))) __bf16 v16bf;
typedef __attribute__((ext_vector_type(8)))  float  v8f;

__device__ __forceinline__ __bf16 f2bf(float f) { return (__bf16)f; }

// A fragment: 16x32 bf16 tile, row-major source, leading dim ld.
// Contiguous k order (same as load_frag_b): lane L holds row M=L&15; K = (L>=16?16:0) + e.
__device__ __forceinline__ v16bf load_frag_a(const __bf16* p, int ld, int lane) {
  v16bf a;
  const int r  = lane & 15;
  const int kb = (lane >= 16) ? 16 : 0;
#pragma unroll
  for (int e = 0; e < 16; ++e) {
    const int k = kb + e;
    a[e] = p[r * ld + k];
  }
  return a;
}

// B fragment: 32x16 bf16 tile (K x N), row-major source, leading dim ld.
// Contiguous k order (same as load_frag_a): lane L holds col N=L&15; K = (L>=16?16:0) + e.
__device__ __forceinline__ v16bf load_frag_b(const __bf16* p, int ld, int lane) {
  v16bf b;
  const int c  = lane & 15;
  const int kb = (lane >= 16) ? 16 : 0;
#pragma unroll
  for (int e = 0; e < 16; ++e) b[e] = p[(kb + e) * ld + c];
  return b;
}

__device__ __forceinline__ v8f wmma_bf16(v16bf a, v16bf b, v8f c) {
  return __builtin_amdgcn_wmma_f32_16x16x32_bf16(false, a, false, b, (short)0, c,
                                                 false, false);
}

// Async global -> LDS 128-bit copy (CDNA5, ASYNCcnt). Inline asm: portable
// across toolchains regardless of the clang builtin's addrspace signature.
__device__ __forceinline__ void async_copy_b128(uint32_t lds_off, const float* g) {
  asm volatile("global_load_async_to_lds_b128 %0, %1, off"
               :: "v"(lds_off), "v"(g) : "memory");
}
__device__ __forceinline__ void wait_async0() {
  asm volatile("s_wait_asynccnt 0" ::: "memory");
}

// ---------------- kernel 1: convert Wg / Wl to bf16 ----------------
__global__ void k_convert_wb(const float* __restrict__ Wg,
                             const float* __restrict__ Wl,
                             __bf16* __restrict__ Wgbf,
                             __bf16* __restrict__ Wlbf) {
  int idx = blockIdx.x * blockDim.x + threadIdx.x;   // 131072 threads
  if (idx < HID_ * HID_)        Wgbf[idx]               = f2bf(Wg[idx]);
  else if (idx < 2 * HID_ * HID_) Wlbf[idx - HID_*HID_] = f2bf(Wl[idx - HID_*HID_]);
}

// ---------------- kernel 2: tiny MLP (x->h0->h1), output h1 in bf16 ----------------
__global__ void k_mlp(const float* __restrict__ x,  const float* __restrict__ W0,
                      const float* __restrict__ b0, const float* __restrict__ W1,
                      const float* __restrict__ b1, __bf16* __restrict__ h1bf) {
  __shared__ float h0s[HID_];
  const int t = threadIdx.x;              // 256 threads, 1 block
  for (int b = 0; b < B_; ++b) {
    float acc = b0[t];
    for (int k = 0; k < LAT_; ++k) acc += x[b * LAT_ + k] * W0[k * HID_ + t];
    h0s[t] = acc > 0.f ? acc : 0.f;
    __syncthreads();
    float acc1 = b1[t];
    for (int k = 0; k < HID_; ++k) acc1 += h0s[k] * W1[k * HID_ + t];
    h1bf[b * HID_ + t] = f2bf(acc1 > 0.f ? acc1 : 0.f);
    __syncthreads();
  }
}

// ---------------- kernel 3: h2 = h1 @ W2 + b2  (M=16, K=256, N=65536) ----------
// Block = 256 threads (8 waves) covering a 128-wide N stripe; W2 tiles are
// staged fp32 into LDS with coalesced async b128 copies (ASYNCcnt), then each
// wave builds its bf16 B fragment from LDS and runs WMMA.
#define TW_ 128
__global__ void k_gemm_h2(const __bf16* __restrict__ h1bf,
                          const float* __restrict__ W2,
                          const float* __restrict__ b2,
                          __bf16* __restrict__ h2bf) {
  __shared__ float tile[32][TW_ + 4];     // 32 x 132 f32, padded vs bank conflicts
  const int tid  = threadIdx.x;
  const int lane = tid & 31;
  const int wave = tid >> 5;              // 8 waves -> 8 x 16-col tiles
  const int nblk = blockIdx.x * TW_;      // 512 blocks
  const int n0   = nblk + wave * 16;

  v8f acc = {};
  for (int kc = 0; kc < HID_ / 32; ++kc) {
    // stage W2[kc*32 .. +31][nblk .. +127] (16 KB) : 4 async b128 per thread
#pragma unroll
    for (int it = 0; it < 4; ++it) {
      const int q   = it * 256 + tid;     // 0..1023 chunks of 4 floats
      const int row = q >> 5;             // 128/4 = 32 chunks per row
      const int c4  = (q & 31) << 2;
      const float* g = W2 + (size_t)(kc * 32 + row) * NHID_ + nblk + c4;
      async_copy_b128((uint32_t)(uintptr_t)&tile[row][c4], g);
    }
    if (kc + 1 < HID_ / 32)               // prefetch next K-slab of the stream
      __builtin_prefetch(W2 + (size_t)(kc + 1) * 32 * NHID_ + nblk + tid * 4, 0, 1);
    wait_async0();
    __syncthreads();

    v16bf a = load_frag_a(h1bf + kc * 32, HID_, lane);
    v16bf b;
    {
      const int c  = lane & 15;
      const int kb = (lane >= 16) ? 16 : 0;
#pragma unroll
      for (int e = 0; e < 16; ++e) b[e] = f2bf(tile[kb + e][wave * 16 + c]);
    }
    acc = wmma_bf16(a, b, acc);
    __syncthreads();
  }

  const int c  = lane & 15;
  const int rb = (lane >= 16) ? 8 : 0;
  const float bias = b2[n0 + c];
#pragma unroll
  for (int r = 0; r < 8; ++r)
    h2bf[(size_t)(rb + r) * NHID_ + n0 + c] = f2bf(acc[r] + bias);  // row = batch
}

// ---------------- kernel 4: gfeat[b] = h2[b](256x256) @ Wg ----------------
__global__ void k_gfeat(const __bf16* __restrict__ h2bf,
                        const __bf16* __restrict__ Wgbf,
                        float* __restrict__ gf,
                        __bf16* __restrict__ gfbf) {
  const int tid  = blockIdx.x * blockDim.x + threadIdx.x;
  const int lane = tid & 31;
  const int wid  = tid >> 5;                 // 4096 waves = 16 batch * 256 tiles
  if (wid >= B_ * 256) return;
  const int b  = wid >> 8;
  const int t  = wid & 255;
  const int m0 = (t >> 4) << 4;
  const int n0 = (t & 15) << 4;
  v8f acc = {};
#pragma unroll
  for (int kc = 0; kc < HID_ / 32; ++kc) {
    v16bf a = load_frag_a(h2bf + (size_t)b * NHID_ + m0 * HID_ + kc * 32, HID_, lane);
    v16bf bb = load_frag_b(Wgbf + kc * 32 * HID_ + n0, HID_, lane);
    acc = wmma_bf16(a, bb, acc);
  }
  const int c  = lane & 15;
  const int rb = (lane >= 16) ? 8 : 0;
#pragma unroll
  for (int r = 0; r < 8; ++r) {
    const size_t o = (size_t)b * NHID_ + (size_t)(m0 + rb + r) * HID_ + n0 + c;
    gf[o]   = acc[r];
    gfbf[o] = f2bf(acc[r]);
  }
}

// ---------------- kernel 5: per-node attention scores a_s, a_d ----------------
__global__ void k_scores(const float* __restrict__ gf,
                         const float* __restrict__ att_src,
                         const float* __restrict__ att_dst,
                         float* __restrict__ as_, float* __restrict__ ad_) {
  const int gid = blockIdx.x * blockDim.x + threadIdx.x;   // 16384 = B*N*H
  if (gid >= B_ * N_ * H_) return;
  const int h = gid & 3, n = (gid >> 2) & 255, b = gid >> 10;
  const float* g = gf + (size_t)b * NHID_ + n * HID_ + h * C_;
  float s = 0.f, d = 0.f;
  for (int c = 0; c < C_; ++c) {
    const float v = g[c];
    s += v * att_src[h * C_ + c];
    d += v * att_dst[h * C_ + c];
  }
  as_[gid] = s;
  ad_[gid] = d;
}

// ---------------- kernel 6: triangular softmax + WMMA aggregate + ELU ----------
// grid = B*H*(N/16) blocks; block = 512 (16 waves).
// Phase 1: wave r builds softmax row j = j0+r of alpha (i<=j) into LDS (bf16).
// Phase 2: waves 0..3 compute 16x64 output tile = alpha(16x256) @ gfeat_h(256x64).
#define ALD 272
__global__ void k_attention(const float* __restrict__ as_,
                            const float* __restrict__ ad_,
                            const __bf16* __restrict__ gfbf,
                            const float* __restrict__ bg,
                            __bf16* __restrict__ h3bf) {
  __shared__ __bf16 alpha[16][ALD];
  const int b  = blockIdx.x >> 6;
  const int h  = (blockIdx.x >> 4) & 3;
  const int j0 = (blockIdx.x & 15) << 4;
  const int wave = threadIdx.x >> 5;
  const int lane = threadIdx.x & 31;

  { // phase 1
    const int j = j0 + wave;
    const float ad = ad_[((b * N_) + j) * H_ + h];
    float ev[8];
    float m = -1e30f;
#pragma unroll
    for (int t = 0; t < 8; ++t) {
      const int i = lane + t * 32;
      float e = -1e30f;
      if (i <= j) {
        const float z = as_[((b * N_) + i) * H_ + h] + ad;
        e = z > 0.f ? z : 0.2f * z;          // leaky_relu(0.2)
      }
      ev[t] = e;
      m = fmaxf(m, e);
    }
    for (int off = 16; off; off >>= 1) m = fmaxf(m, __shfl_xor(m, off));
    float s = 0.f;
#pragma unroll
    for (int t = 0; t < 8; ++t) {
      const int i = lane + t * 32;
      const float p = (i <= j) ? expf(ev[t] - m) : 0.f;
      ev[t] = p;
      s += p;
    }
    for (int off = 16; off; off >>= 1) s += __shfl_xor(s, off);
    const float inv = 1.f / s;
#pragma unroll
    for (int t = 0; t < 8; ++t) alpha[wave][lane + t * 32] = f2bf(ev[t] * inv);
  }
  __syncthreads();

  if (wave < 4) { // phase 2: out cols n0..n0+15 within this head's 64 channels
    const int n0 = wave << 4;
    v8f acc = {};
#pragma unroll
    for (int kc = 0; kc < N_ / 32; ++kc) {
      v16bf a  = load_frag_a(&alpha[0][0] + kc * 32, ALD, lane);
      v16bf bb = load_frag_b(gfbf + (size_t)b * NHID_ + (size_t)(kc * 32) * HID_
                             + h * C_ + n0, HID_, lane);
      acc = wmma_bf16(a, bb, acc);
    }
    const int c  = lane & 15;
    const int rb = (lane >= 16) ? 8 : 0;
    const int ch = h * C_ + n0 + c;
    const float bias = bg[ch];
#pragma unroll
    for (int r = 0; r < 8; ++r) {
      const int j = j0 + rb + r;
      float v = acc[r] + bias;
      v = v > 0.f ? v : (expf(v) - 1.f);     // ELU
      h3bf[(size_t)b * NHID_ + (size_t)j * HID_ + ch] = f2bf(v);
    }
  }
}

// ---------------- kernel 7: logits[b] = h3[b] @ Wl + bl ----------------
__global__ void k_logits(const __bf16* __restrict__ h3bf,
                         const __bf16* __restrict__ Wlbf,
                         const float* __restrict__ bl,
                         float* __restrict__ logits) {
  const int tid  = blockIdx.x * blockDim.x + threadIdx.x;
  const int lane = tid & 31;
  const int wid  = tid >> 5;                 // 4096 waves
  if (wid >= B_ * 256) return;
  const int b  = wid >> 8;
  const int t  = wid & 255;
  const int m0 = (t >> 4) << 4;
  const int n0 = (t & 15) << 4;
  v8f acc = {};
#pragma unroll
  for (int kc = 0; kc < HID_ / 32; ++kc) {
    v16bf a  = load_frag_a(h3bf + (size_t)b * NHID_ + m0 * HID_ + kc * 32, HID_, lane);
    v16bf bb = load_frag_b(Wlbf + kc * 32 * HID_ + n0, HID_, lane);
    acc = wmma_bf16(a, bb, acc);
  }
  const int c  = lane & 15;
  const int rb = (lane >= 16) ? 8 : 0;
  const float bias = bl[n0 + c];
#pragma unroll
  for (int r = 0; r < 8; ++r)
    logits[(size_t)b * (N_ * N_) + (size_t)(m0 + rb + r) * N_ + n0 + c] = acc[r] + bias;
}

// ---------------- kernel 8: zero output ----------------
__global__ void k_zero(float* __restrict__ out, int n) {
  const int idx = blockIdx.x * blockDim.x + threadIdx.x;
  if (idx < n) out[idx] = 0.f;
}

// ---------------- kernel 9: hard gumbel = row argmax; adj = clip(y + y^T) ------
// One wave per (b, row). y is exactly one-hot(argmax(logits+gumbel)).
__global__ void k_argmax_scatter(const float* __restrict__ logits,
                                 const float* __restrict__ gumbel,
                                 float* __restrict__ out) {
  const int tid  = blockIdx.x * blockDim.x + threadIdx.x;
  const int lane = tid & 31;
  const int wid  = tid >> 5;                 // 4096 waves = B*N rows
  if (wid >= B_ * N_) return;
  const int b = wid >> 8;
  const int r = wid & 255;
  const size_t base = (size_t)b * (N_ * N_) + (size_t)r * N_;
  float best = -3.4e38f;
  int   bidx = 0;
  for (int i = lane; i < N_; i += 32) {
    const float v = logits[base + i] + gumbel[base + i];
    if (v > best || (v == best && i < bidx)) { best = v; bidx = i; }
  }
  for (int off = 16; off; off >>= 1) {
    const float ov = __shfl_xor(best, off);
    const int   oi = __shfl_xor(bidx, off);
    if (ov > best || (ov == best && oi < bidx)) { best = ov; bidx = oi; }
  }
  if (lane == 0) {
    out[base + bidx] = 1.f;                                        // y
    out[(size_t)b * (N_ * N_) + (size_t)bidx * N_ + r] = 1.f;      // + y^T, clipped
  }
}

// ---------------- host launch ----------------
extern "C" void kernel_launch(void* const* d_in, const int* in_sizes, int n_in,
                              void* d_out, int out_size, void* d_ws, size_t ws_size,
                              hipStream_t stream) {
  const float* x       = (const float*)d_in[0];
  const float* W0      = (const float*)d_in[1];
  const float* b0      = (const float*)d_in[2];
  const float* W1      = (const float*)d_in[3];
  const float* b1      = (const float*)d_in[4];
  const float* W2      = (const float*)d_in[5];
  const float* b2      = (const float*)d_in[6];
  const float* Wg      = (const float*)d_in[7];
  const float* att_src = (const float*)d_in[8];
  const float* att_dst = (const float*)d_in[9];
  const float* bg      = (const float*)d_in[10];
  const float* Wl      = (const float*)d_in[11];
  const float* bl      = (const float*)d_in[12];
  const float* gumbel  = (const float*)d_in[13];
  float* out = (float*)d_out;

  // workspace layout (all offsets 256B aligned)
  char* ws = (char*)d_ws;
  size_t off = 0;
  __bf16* h1bf  = (__bf16*)(ws + off); off += (size_t)B_ * HID_ * 2;          // 8 KB
  __bf16* Wgbf  = (__bf16*)(ws + off); off += (size_t)HID_ * HID_ * 2;        // 128 KB
  __bf16* Wlbf  = (__bf16*)(ws + off); off += (size_t)HID_ * HID_ * 2;        // 128 KB
  __bf16* h2bf  = (__bf16*)(ws + off); off += (size_t)B_ * NHID_ * 2;         // 2 MB
  float*  gf    = (float*) (ws + off); off += (size_t)B_ * NHID_ * 4;         // 4 MB
  __bf16* gfbf  = (__bf16*)(ws + off); off += (size_t)B_ * NHID_ * 2;         // 2 MB
  float*  as_   = (float*) (ws + off); off += (size_t)B_ * N_ * H_ * 4;       // 64 KB
  float*  ad_   = (float*) (ws + off); off += (size_t)B_ * N_ * H_ * 4;       // 64 KB
  __bf16* h3bf  = (__bf16*)(ws + off); off += (size_t)B_ * NHID_ * 2;         // 2 MB
  float*  logit = (float*) (ws + off); off += (size_t)B_ * N_ * N_ * 4;       // 4 MB

  k_convert_wb<<<512, 256, 0, stream>>>(Wg, Wl, Wgbf, Wlbf);
  k_mlp<<<1, 256, 0, stream>>>(x, W0, b0, W1, b1, h1bf);
  k_gemm_h2<<<NHID_ / TW_, 256, 0, stream>>>(h1bf, W2, b2, h2bf);  // 512 blocks
  k_gfeat<<<512, 256, 0, stream>>>(h2bf, Wgbf, gf, gfbf);          // 4096 waves
  k_scores<<<64, 256, 0, stream>>>(gf, att_src, att_dst, as_, ad_);
  k_attention<<<B_ * H_ * (N_ / 16), 512, 0, stream>>>(as_, ad_, gfbf, bg, h3bf);
  k_logits<<<512, 256, 0, stream>>>(h3bf, Wlbf, bl, logit);        // 4096 waves
  k_zero<<<(out_size + 255) / 256, 256, 0, stream>>>(out, out_size);
  k_argmax_scatter<<<512, 256, 0, stream>>>(logit, gumbel, out);   // 4096 waves
}
